// MIGC_plus_66305705115727
// MI455X (gfx1250) — matrix operand order repaired and checked by reading, not compiled, not run on hardware
//
#include <hip/hip_runtime.h>
#include <hip/hip_bf16.h>
#include <math.h>

// ---------- types ----------
typedef __attribute__((ext_vector_type(16))) _Float16 v16h;
typedef __attribute__((ext_vector_type(8)))  _Float16 v8h;
typedef __attribute__((ext_vector_type(8)))  float    v8f;

#define CC   1280
#define HEADS 8
#define HD   160
#define HWSZ 4096
#define INST 6
#define FULLR 512
#define NUMPRO 30

__device__ inline v8f v8f_zero() {
  v8f z;
#pragma unroll
  for (int i = 0; i < 8; ++i) z[i] = 0.0f;
  return z;
}

// CDNA5 async global->LDS DMA (ASYNCcnt-tracked, no VGPR staging).
// lds_off is a byte offset within the workgroup's dynamic LDS (base 0).
__device__ inline void async_b128(const _Float16* g, unsigned lds_off) {
  asm volatile("global_load_async_to_lds_b128 %0, %1, off"
               :: "v"(lds_off), "v"((unsigned long long)g)
               : "memory");
}

// ---------- bilinear resize 512x512 -> 64x64 ----------
__global__ void k_resize(const float* __restrict__ in, float* __restrict__ out, int nch) {
  int idx = blockIdx.x * blockDim.x + threadIdx.x;
  int total = nch * 64 * 64;
  if (idx >= total) return;
  int ox = idx & 63, oy = (idx >> 6) & 63, ch = idx >> 12;
  float cy = fminf(fmaxf((oy + 0.5f) * 8.0f - 0.5f, 0.0f), 511.0f);
  float cx = fminf(fmaxf((ox + 0.5f) * 8.0f - 0.5f, 0.0f), 511.0f);
  int ylo = (int)floorf(cy); int yhi = min(ylo + 1, 511); float ty = cy - ylo;
  int xlo = (int)floorf(cx); int xhi = min(xlo + 1, 511); float tx = cx - xlo;
  const float* p = in + (size_t)ch * FULLR * FULLR;
  float r0 = p[ylo * FULLR + xlo] * (1.f - tx) + p[ylo * FULLR + xhi] * tx;
  float r1 = p[yhi * FULLR + xlo] * (1.f - tx) + p[yhi * FULLR + xhi] * tx;
  out[idx] = r0 * (1.f - ty) + r1 * ty;
}

// ---------- mig = sum_i ca_x[i]*gms[i] / (sum_i gms[i] + 1e-6) ----------
__global__ void k_mig(const float* __restrict__ ca_x, const float* __restrict__ gm,
                      float* __restrict__ out) {
  int idx = blockIdx.x * blockDim.x + threadIdx.x;
  if (idx >= HWSZ * CC) return;
  int hw = idx / CC;
  float num = ca_x[idx];   // channel 0 weight 1.0
  float den = 1.0f;
#pragma unroll
  for (int i = 0; i < INST; ++i) {
    float w = gm[i * HWSZ + hw] * 10.0f;
    num += ca_x[(size_t)(i + 1) * HWSZ * CC + idx] * w;
    den += w;
  }
  out[idx] = num / (den + 1e-6f);
}

// ---------- pos_net: box(6,4) -> (6,768) ----------
__global__ __launch_bounds__(256) void k_posnet(
    const float* __restrict__ box, const float* __restrict__ w1, const float* __restrict__ b1,
    const float* __restrict__ w2, const float* __restrict__ b2, const float* __restrict__ w3,
    const float* __restrict__ b3, float* __restrict__ boxtok) {
  __shared__ float emb[64], h1[512], h2[512];
  int bi = blockIdx.x, tid = threadIdx.x;
  if (tid < 64) {
    int f = tid >> 3, s = (tid >> 2) & 1, bc = tid & 3;
    float freq = powf(100.0f, (float)f / 8.0f);
    float xb = box[bi * 4 + bc] * freq;
    emb[tid] = s ? cosf(xb) : sinf(xb);
  }
  __syncthreads();
  for (int o = tid; o < 512; o += 256) {
    float a = b1[o];
    for (int k = 0; k < 64; ++k) a += emb[k] * w1[o * 64 + k];
    h1[o] = a / (1.0f + __expf(-a));
  }
  __syncthreads();
  for (int o = tid; o < 512; o += 256) {
    float a = b2[o];
    for (int k = 0; k < 512; ++k) a += h1[k] * w2[o * 512 + k];
    h2[o] = a / (1.0f + __expf(-a));
  }
  __syncthreads();
  for (int o = tid; o < 768; o += 256) {
    float a = b3[o];
    for (int k = 0; k < 512; ++k) a += h2[k] * w3[o * 512 + k];
    boxtok[bi * 768 + o] = a;
  }
}

// ---------- EA k/v : ctx(6,2,768) @ W(1280,768)^T ----------
__global__ void k_kv_ea(const float* __restrict__ pooler, const float* __restrict__ boxtok,
                        const float* __restrict__ kw, const float* __restrict__ vw,
                        float* __restrict__ kea, float* __restrict__ vea) {
  int idx = blockIdx.x * blockDim.x + threadIdx.x;
  if (idx >= INST * 2 * CC) return;
  int c = idx % CC, t = (idx / CC) & 1, bi = idx / (2 * CC);
  const float* ctx = (t == 0) ? (pooler + (size_t)(1 + bi) * 768) : (boxtok + (size_t)bi * 768);
  float ka = 0.f, va = 0.f;
  for (int k = 0; k < 768; ++k) {
    float x = ctx[k];
    ka += x * kw[(size_t)c * 768 + k];
    va += x * vw[(size_t)c * 768 + k];
  }
  kea[idx] = ka; vea[idx] = va;
}

// ---------- LayerNorm rows of 1280, write f16 ----------
__global__ __launch_bounds__(256) void k_ln_f16(const float* __restrict__ x,
                                                const float* __restrict__ w,
                                                const float* __restrict__ b,
                                                _Float16* __restrict__ y) {
  __shared__ float red[256];
  int row = blockIdx.x, tid = threadIdx.x;
  const float* xr = x + (size_t)row * CC;
  float s = 0.f;
  for (int c = tid; c < CC; c += 256) s += xr[c];
  red[tid] = s; __syncthreads();
  for (int st = 128; st > 0; st >>= 1) { if (tid < st) red[tid] += red[tid + st]; __syncthreads(); }
  float mean = red[0] * (1.0f / CC);
  __syncthreads();
  float s2 = 0.f;
  for (int c = tid; c < CC; c += 256) { float d = xr[c] - mean; s2 += d * d; }
  red[tid] = s2; __syncthreads();
  for (int st = 128; st > 0; st >>= 1) { if (tid < st) red[tid] += red[tid + st]; __syncthreads(); }
  float inv = rsqrtf(red[0] * (1.0f / CC) + 1e-5f);
  for (int c = tid; c < CC; c += 256)
    y[(size_t)row * CC + c] = (_Float16)((xr[c] - mean) * inv * w[c] + b[c]);
}

// ---------- f32 -> f16 convert ----------
__global__ void k_cvt_h(const float* __restrict__ x, _Float16* __restrict__ y, size_t n) {
  size_t i = (size_t)blockIdx.x * blockDim.x + threadIdx.x;
  if (i < n) y[i] = (_Float16)x[i];
}

// ---------- WMMA GEMM: D[M,N] = A[M,K] * W[N,K]^T (+bias), f16 in, f32 acc ----------
// 64x64 block tile, 8 waves each own 16 rows x 32 cols (2 accumulators).
// Global->LDS staging uses CDNA5 async DMA with double buffering:
//   dynamic LDS layout: A buffers [0,8192), B buffers [8192,16384), 4096 B each.
__global__ __launch_bounds__(256) void k_gemm_wmma(
    const _Float16* __restrict__ A, const _Float16* __restrict__ Bw,
    const float* __restrict__ bias, void* __restrict__ out,
    int M, int N, int K, int out_half) {
  extern __shared__ char smem[];
  const int tid = threadIdx.x, lane = tid & 31, wave = tid >> 5;
  const int tm = wave & 3;        // m tile (16 rows)
  const int tn = wave >> 2;       // n half (32 cols)
  const int bm = blockIdx.y * 64, bn = blockIdx.x * 64;
  const int lr = tid >> 2, lcg = (tid & 3) * 8;   // cooperative load mapping
  const unsigned ldst = (unsigned)(lr * 64 + lcg * 2);

  v8f acc[2];
  acc[0] = v8f_zero(); acc[1] = v8f_zero();

  // prologue: stage 0 -> buffer 0
  async_b128(A  + (size_t)(bm + lr) * K + lcg, 0u + ldst);
  async_b128(Bw + (size_t)(bn + lr) * K + lcg, 8192u + ldst);

  const int NSTEP = K / 32;
  for (int s = 0; s < NSTEP; ++s) {
    const int cur = s & 1;
    if (s + 1 < NSTEP) {
      const int kk = (s + 1) * 32;
      const unsigned nb = (unsigned)((s + 1) & 1) * 4096u;
      async_b128(A  + (size_t)(bm + lr) * K + kk + lcg, nb + ldst);
      async_b128(Bw + (size_t)(bn + lr) * K + kk + lcg, 8192u + nb + ldst);
      asm volatile("s_wait_asynccnt 0x2" ::: "memory"); // stage s done, s+1 in flight
    } else {
      asm volatile("s_wait_asynccnt 0x0" ::: "memory");
    }
    __syncthreads();
    const char* baseA = smem + cur * 4096;
    const char* baseB = smem + 8192 + cur * 4096;
    int am = tm * 16 + (lane & 15);
    v8h alo = *(const v8h*)(baseA + am * 64 + (lane >> 4) * 16);
    v8h ahi = *(const v8h*)(baseA + am * 64 + (lane >> 4) * 16 + 32);
    v16h af;
#pragma unroll
    for (int j = 0; j < 8; ++j) { af[j] = alo[j]; af[j + 8] = ahi[j]; }
#pragma unroll
    for (int sn = 0; sn < 2; ++sn) {
      int bnr = tn * 32 + sn * 16 + (lane & 15);
      v8h blo = *(const v8h*)(baseB + bnr * 64 + (lane >> 4) * 16);
      v8h bhi = *(const v8h*)(baseB + bnr * 64 + (lane >> 4) * 16 + 32);
      v16h bf;
#pragma unroll
      for (int j = 0; j < 8; ++j) { bf[j] = blo[j]; bf[j + 8] = bhi[j]; }
      acc[sn] = __builtin_amdgcn_wmma_f32_16x16x32_f16(false, af, false, bf, (short)0,
                                                       acc[sn], false, false);
    }
    __syncthreads();
  }
#pragma unroll
  for (int sn = 0; sn < 2; ++sn) {
    int col = bn + tn * 32 + sn * 16 + (lane & 15);
    float bv = bias ? bias[col] : 0.0f;
#pragma unroll
    for (int j = 0; j < 8; ++j) {
      int row = bm + tm * 16 + (lane >> 4) * 8 + j;
      float v = acc[sn][j] + bv;
      if (out_half) ((_Float16*)out)[(size_t)row * N + col] = (_Float16)v;
      else          ((float*)out)[(size_t)row * N + col] = v;
    }
  }
}

// ---------- EA attention: 2 keys per (bi,head,query) ----------
__global__ void k_ea_attn(const float* __restrict__ q, const float* __restrict__ kea,
                          const float* __restrict__ vea, _Float16* __restrict__ ao) {
  int idx = blockIdx.x * blockDim.x + threadIdx.x;
  if (idx >= INST * HWSZ * HEADS) return;
  int h = idx % HEADS, qi = (idx / HEADS) % HWSZ, bi = idx / (HEADS * HWSZ);
  size_t base = ((size_t)bi * HWSZ + qi) * CC + h * HD;
  const float* k0 = kea + ((size_t)bi * 2 + 0) * CC + h * HD;
  const float* k1 = kea + ((size_t)bi * 2 + 1) * CC + h * HD;
  float s0 = 0.f, s1 = 0.f;
  for (int j = 0; j < HD; ++j) { float x = q[base + j]; s0 += x * k0[j]; s1 += x * k1[j]; }
  const float scale = 0.07905694150420949f; // 160^-0.5
  s0 *= scale; s1 *= scale;
  float m = fmaxf(s0, s1);
  float p0 = __expf(s0 - m), p1 = __expf(s1 - m);
  float inv = 1.0f / (p0 + p1);
  p0 *= inv; p1 *= inv;
  const float* v0 = vea + ((size_t)bi * 2 + 0) * CC + h * HD;
  const float* v1 = vea + ((size_t)bi * 2 + 1) * CC + h * HD;
  for (int j = 0; j < HD; ++j) ao[base + j] = (_Float16)(p0 * v0[j] + p1 * v1[j]);
}

// ---------- ea *= gm ----------
__global__ void k_scale_gm(float* __restrict__ shading, const float* __restrict__ gm) {
  size_t idx = (size_t)blockIdx.x * blockDim.x + threadIdx.x;
  if (idx >= (size_t)INST * HWSZ * CC) return;
  shading[idx] *= gm[idx / CC];
}

// ---------- fused LA flash attention with WMMA ----------
__global__ __launch_bounds__(256) void k_flash_la(
    const _Float16* __restrict__ lq, const _Float16* __restrict__ lk,
    const _Float16* __restrict__ lv, const float* __restrict__ gm,
    const float* __restrict__ sup, _Float16* __restrict__ outh) {
  __shared__ __align__(32) _Float16 Qs[64][160];
  __shared__ __align__(32) _Float16 Ks[32][160];
  __shared__ __align__(32) _Float16 VTs[160][32];
  __shared__ __align__(32) _Float16 Ps[4][16][32];
  __shared__ float gq[7][64];
  __shared__ float gk[7][32];
  __shared__ float mrow[64], lrow[64], arow[64];
  __shared__ float pmax[2][64], psum[2][64];

  const int tid = threadIdx.x, lane = tid & 31, wave = tid >> 5;
  const int ws = wave & 3;    // query row strip (16 rows)
  const int half = wave >> 2; // key/dim half
  const int q0 = blockIdx.x * 64;
  const int head = blockIdx.y;
  const int hc = head * HD;
  const float scale = 0.07905694150420949f;

  for (int i = tid; i < 64 * HD; i += 256) {
    int r = i / HD, d = i % HD;
    Qs[r][d] = lq[(size_t)(q0 + r) * CC + hc + d];
  }
  for (int i = tid; i < 7 * 64; i += 256) {
    int ch = i / 64, r = i % 64;
    gq[ch][r] = (ch < 6) ? gm[ch * HWSZ + q0 + r] : sup[q0 + r];
  }
  if (tid < 64) { mrow[tid] = -3.0e38f; lrow[tid] = 0.0f; }

  v8f accO[5];
#pragma unroll
  for (int i = 0; i < 5; ++i) accO[i] = v8f_zero();

  const int row0 = ws * 16 + (lane >> 4) * 8;
  const int kloc = half * 16 + (lane & 15);

  for (int kb0 = 0; kb0 < HWSZ; kb0 += 32) {
    __syncthreads();
    for (int i = tid; i < 32 * HD; i += 256) {
      int r = i / HD, d = i % HD;
      Ks[r][d]  = lk[(size_t)(kb0 + r) * CC + hc + d];
      VTs[d][r] = lv[(size_t)(kb0 + r) * CC + hc + d];
    }
    if (tid < 7 * 32) {
      int ch = tid / 32, r = tid % 32;
      gk[ch][r] = (ch < 6) ? gm[ch * HWSZ + kb0 + r] : sup[kb0 + r];
    }
    __syncthreads();

    // S strip (16 rows x 16 keys per wave)
    v8f sacc = v8f_zero();
#pragma unroll
    for (int ks = 0; ks < HD; ks += 32) {
      int am = ws * 16 + (lane & 15);
      int akb = (lane >> 4) * 8;
      v8h alo = *(const v8h*)&Qs[am][ks + akb];
      v8h ahi = *(const v8h*)&Qs[am][ks + akb + 16];
      v16h af;
#pragma unroll
      for (int j = 0; j < 8; ++j) { af[j] = alo[j]; af[j + 8] = ahi[j]; }
      int bkey = half * 16 + (lane & 15);
      v8h blo = *(const v8h*)&Ks[bkey][ks + akb];
      v8h bhi = *(const v8h*)&Ks[bkey][ks + akb + 16];
      v16h bf;
#pragma unroll
      for (int j = 0; j < 8; ++j) { bf[j] = blo[j]; bf[j + 8] = bhi[j]; }
      sacc = __builtin_amdgcn_wmma_f32_16x16x32_f16(false, af, false, bf, (short)0, sacc, false, false);
    }
    // mask + scale
    float sv[8];
#pragma unroll
    for (int j = 0; j < 8; ++j) {
      int r = row0 + j;
      float md = 0.f;
#pragma unroll
      for (int ch = 0; ch < 7; ++ch) md += gq[ch][r] * gk[ch][kloc];
      sv[j] = (md > 0.f) ? sacc[j] * scale : -1.0e9f;
    }
    // row max across the 16-lane group
    float rmax[8];
#pragma unroll
    for (int j = 0; j < 8; ++j) {
      float v = sv[j];
      v = fmaxf(v, __shfl_xor(v, 1, 32));
      v = fmaxf(v, __shfl_xor(v, 2, 32));
      v = fmaxf(v, __shfl_xor(v, 4, 32));
      v = fmaxf(v, __shfl_xor(v, 8, 32));
      rmax[j] = v;
    }
    if ((lane & 15) == 0) {
#pragma unroll
      for (int j = 0; j < 8; ++j) pmax[half][row0 + j] = rmax[j];
    }
    __syncthreads();
    if (tid < 64) {
      float mb = fmaxf(pmax[0][tid], pmax[1][tid]);
      float mo = mrow[tid];
      float mn = fmaxf(mo, mb);
      arow[tid] = __expf(mo - mn);
      mrow[tid] = mn;
    }
    __syncthreads();
#pragma unroll
    for (int j = 0; j < 8; ++j) {
      int r = row0 + j;
      float p = __expf(sv[j] - mrow[r]);
      sv[j] = p;
      Ps[ws][(lane >> 4) * 8 + j][kloc] = (_Float16)p;
    }
    float rsum[8];
#pragma unroll
    for (int j = 0; j < 8; ++j) {
      float v = sv[j];
      v += __shfl_xor(v, 1, 32);
      v += __shfl_xor(v, 2, 32);
      v += __shfl_xor(v, 4, 32);
      v += __shfl_xor(v, 8, 32);
      rsum[j] = v;
    }
    if ((lane & 15) == 0) {
#pragma unroll
      for (int j = 0; j < 8; ++j) psum[half][row0 + j] = rsum[j];
    }
    __syncthreads();
    if (tid < 64) lrow[tid] = lrow[tid] * arow[tid] + psum[0][tid] + psum[1][tid];
    // P (16x32) * V (32x160): each wave does its 80-dim half (5 n-tiles)
    {
      int am = lane & 15, akb = (lane >> 4) * 8;
      v8h plo = *(const v8h*)&Ps[ws][am][akb];
      v8h phi = *(const v8h*)&Ps[ws][am][akb + 16];
      v16h pf;
#pragma unroll
      for (int j = 0; j < 8; ++j) { pf[j] = plo[j]; pf[j + 8] = phi[j]; }
#pragma unroll
      for (int nt = 0; nt < 5; ++nt) {
#pragma unroll
        for (int j = 0; j < 8; ++j) accO[nt][j] *= arow[row0 + j];
        int dcol = half * 80 + nt * 16 + (lane & 15);
        v8h vlo = *(const v8h*)&VTs[dcol][akb];
        v8h vhi = *(const v8h*)&VTs[dcol][akb + 16];
        v16h vf;
#pragma unroll
        for (int j = 0; j < 8; ++j) { vf[j] = vlo[j]; vf[j + 8] = vhi[j]; }
        accO[nt] = __builtin_amdgcn_wmma_f32_16x16x32_f16(false, pf, false, vf, (short)0, accO[nt], false, false);
      }
    }
  }
  __syncthreads();
#pragma unroll
  for (int nt = 0; nt < 5; ++nt) {
    int dcol = hc + half * 80 + nt * 16 + (lane & 15);
#pragma unroll
    for (int j = 0; j < 8; ++j) {
      int r = row0 + j;
      outh[(size_t)(q0 + r) * CC + dcol] = (_Float16)(accO[nt][j] / lrow[r]);
    }
  }
}

// ---------- SAC: per-(p,hw) channel dot ----------
__global__ __launch_bounds__(256) void k_scale8(const float* __restrict__ shading,
                                                const float* __restrict__ w,
                                                const float* __restrict__ bconst,
                                                float* __restrict__ scale8) {
  int gw = (blockIdx.x * 256 + threadIdx.x) >> 5;
  int lane = threadIdx.x & 31;
  if (gw >= 8 * HWSZ) return;
  int p = gw >> 12, hw = gw & (HWSZ - 1);
  float a = 0.f;
  if (p < 7) {
    const float* row = shading + ((size_t)p * HWSZ + hw) * CC;
    for (int c = lane; c < CC; c += 32) a += row[c] * w[c];
  }
  for (int m = 16; m; m >>= 1) a += __shfl_xor(a, m, 32);
  if (lane == 0) scale8[p * HWSZ + hw] = a + bconst[0];
}

__global__ void k_build30(const float* __restrict__ scale8, const int* __restrict__ order,
                          float* __restrict__ scale30) {
  int idx = blockIdx.x * blockDim.x + threadIdx.x;
  if (idx >= NUMPRO * HWSZ) return;
  int hw = idx & (HWSZ - 1), ch = idx >> 12;
  int src;
  if (ch == NUMPRO - 1) src = 6;
  else { int o = order[ch]; src = (o < 6) ? o : 7; }
  scale30[idx] = scale8[src * HWSZ + hw];
}

__global__ __launch_bounds__(256) void k_chanreduce(const float* __restrict__ scale30,
                                                    float* __restrict__ chmean,
                                                    float* __restrict__ chmax) {
  __shared__ float rs[256], rm[256];
  int ch = blockIdx.x, tid = threadIdx.x;
  const float* p = scale30 + (size_t)ch * HWSZ;
  float s = 0.f, m = -3.0e38f;
  for (int i = tid; i < HWSZ; i += 256) { float v = p[i]; s += v; m = fmaxf(m, v); }
  rs[tid] = s; rm[tid] = m; __syncthreads();
  for (int st = 128; st > 0; st >>= 1) {
    if (tid < st) { rs[tid] += rs[tid + st]; rm[tid] = fmaxf(rm[tid], rm[tid + st]); }
    __syncthreads();
  }
  if (tid == 0) { chmean[ch] = rs[0] * (1.0f / HWSZ); chmax[ch] = rm[0]; }
}

__global__ void k_cbam_mlp(const float* __restrict__ chmean, const float* __restrict__ chmax,
                           const float* __restrict__ w1, const float* __restrict__ b1,
                           const float* __restrict__ w2, const float* __restrict__ b2,
                           float* __restrict__ casig) {
  __shared__ float h1m[NUMPRO], h1x[NUMPRO];
  int t = threadIdx.x;
  if (t < NUMPRO) {
    float am = b1[t], ax = b1[t];
    for (int k = 0; k < NUMPRO; ++k) { am += chmean[k] * w1[t * NUMPRO + k]; ax += chmax[k] * w1[t * NUMPRO + k]; }
    h1m[t] = fmaxf(am, 0.f); h1x[t] = fmaxf(ax, 0.f);
  }
  __syncthreads();
  if (t < NUMPRO) {
    float a = 2.0f * b2[t];
    for (int k = 0; k < NUMPRO; ++k) a += (h1m[k] + h1x[k]) * w2[t * NUMPRO + k];
    casig[t] = 1.0f / (1.0f + __expf(-a));
  }
}

__global__ void k_xca_comp(const float* __restrict__ scale30, const float* __restrict__ casig,
                           float* __restrict__ x_ca, float* __restrict__ compmax,
                           float* __restrict__ compmean) {
  int hw = blockIdx.x * blockDim.x + threadIdx.x;
  if (hw >= HWSZ) return;
  float mx = -3.0e38f, sm = 0.f;
  for (int ch = 0; ch < NUMPRO; ++ch) {
    float v = scale30[ch * HWSZ + hw] * casig[ch];
    x_ca[ch * HWSZ + hw] = v;
    mx = fmaxf(mx, v); sm += v;
  }
  compmax[hw] = mx; compmean[hw] = sm * (1.0f / NUMPRO);
}

__global__ void k_conv7(const float* __restrict__ compmax, const float* __restrict__ compmean,
                        const float* __restrict__ w, const float* __restrict__ bnw,
                        const float* __restrict__ bnb, float* __restrict__ sigy) {
  int hw = blockIdx.x * blockDim.x + threadIdx.x;
  if (hw >= HWSZ) return;
  int ox = hw & 63, oy = hw >> 6;
  float a = 0.f;
  for (int ky = 0; ky < 7; ++ky) {
    int iy = oy + ky - 3; if (iy < 0 || iy > 63) continue;
    for (int kx = 0; kx < 7; ++kx) {
      int ix = ox + kx - 3; if (ix < 0 || ix > 63) continue;
      int ii = iy * 64 + ix;
      a += compmax[ii] * w[ky * 7 + kx] + compmean[ii] * w[49 + ky * 7 + kx];
    }
  }
  float y = a * rsqrtf(1.0f + 1e-5f) * bnw[0] + bnb[0];
  sigy[hw] = 1.0f / (1.0f + __expf(-y));
}

__global__ void k_inv(const int* __restrict__ order, int* __restrict__ inv) {
  int t = threadIdx.x;
  if (t < NUMPRO - 1) inv[order[t]] = t;
}

__global__ void k_wfin(const float* __restrict__ x_ca, const float* __restrict__ sigy,
                       const int* __restrict__ inv, float* __restrict__ wfin) {
  int hw = blockIdx.x * blockDim.x + threadIdx.x;
  if (hw >= HWSZ) return;
  float v[7];
  float sy = sigy[hw];
  for (int p = 0; p < 6; ++p) v[p] = x_ca[inv[p] * HWSZ + hw] * sy;
  v[6] = x_ca[(NUMPRO - 1) * HWSZ + hw] * sy;
  float m = v[0];
  for (int p = 1; p < 7; ++p) m = fmaxf(m, v[p]);
  float s = 0.f;
  for (int p = 0; p < 7; ++p) { v[p] = __expf(v[p] - m); s += v[p]; }
  float is = 1.0f / s;
  for (int p = 0; p < 7; ++p) wfin[p * HWSZ + hw] = v[p] * is;
}

__global__ void k_final(const float* __restrict__ shading, const float* __restrict__ wfin,
                        const float* __restrict__ gamma, float* __restrict__ out) {
  int idx = blockIdx.x * blockDim.x + threadIdx.x;
  if (idx >= HWSZ * CC) return;
  int hw = idx / CC;
  float a = 0.f;
#pragma unroll
  for (int p = 0; p < 7; ++p)
    a += shading[(size_t)p * HWSZ * CC + idx] * wfin[p * HWSZ + hw];
  out[idx] = out[idx] + a * tanhf(gamma[0]);
}

// ======================================================================
extern "C" void kernel_launch(void* const* d_in, const int* in_sizes, int n_in,
                              void* d_out, int out_size, void* d_ws, size_t ws_size,
                              hipStream_t stream) {
  const float* ca_x    = (const float*)d_in[0];
  const float* guid    = (const float*)d_in[1];
  const float* supm    = (const float*)d_in[2];
  const float* itok    = (const float*)d_in[3];
  const float* pooler  = (const float*)d_in[4];
  const float* box     = (const float*)d_in[5];
  const float* ln_w    = (const float*)d_in[6];
  const float* ln_b    = (const float*)d_in[7];
  const float* ea_q_w  = (const float*)d_in[8];
  const float* ea_k_w  = (const float*)d_in[9];
  const float* ea_v_w  = (const float*)d_in[10];
  const float* ea_o_w  = (const float*)d_in[11];
  const float* ea_o_b  = (const float*)d_in[12];
  const float* la_q_w  = (const float*)d_in[13];
  const float* la_k_w  = (const float*)d_in[14];
  const float* la_v_w  = (const float*)d_in[15];
  const float* la_o_w  = (const float*)d_in[16];
  const float* la_o_b  = (const float*)d_in[17];
  const float* pn_w1   = (const float*)d_in[18];
  const float* pn_b1   = (const float*)d_in[19];
  const float* pn_w2   = (const float*)d_in[20];
  const float* pn_b2   = (const float*)d_in[21];
  const float* pn_w3   = (const float*)d_in[22];
  const float* pn_b3   = (const float*)d_in[23];
  const float* sac_w   = (const float*)d_in[24];
  const float* sac_b   = (const float*)d_in[25];
  const float* cb_w1   = (const float*)d_in[26];
  const float* cb_b1   = (const float*)d_in[27];
  const float* cb_w2   = (const float*)d_in[28];
  const float* cb_b2   = (const float*)d_in[29];
  const float* cb_scw  = (const float*)d_in[30];
  const float* cb_bnw  = (const float*)d_in[31];
  const float* cb_bnb  = (const float*)d_in[32];
  const float* gamma   = (const float*)d_in[33];
  const int*   order   = (const int*)d_in[36];

  // ---- workspace carve ----
  char* W = (char*)d_ws;
  size_t off = 0;
  auto carve = [&](size_t bytes) -> void* {
    void* r = W + off; off += (bytes + 255) & ~(size_t)255; return r;
  };
  float*     gm      = (float*)carve((size_t)INST * HWSZ * 4);
  float*     sup     = (float*)carve((size_t)HWSZ * 4);
  float*     boxtok  = (float*)carve((size_t)INST * 768 * 4);
  float*     kea     = (float*)carve((size_t)INST * 2 * CC * 4);
  float*     vea     = (float*)carve((size_t)INST * 2 * CC * 4);
  _Float16*  w_eaq   = (_Float16*)carve((size_t)CC * CC * 2);
  _Float16*  w_eao   = (_Float16*)carve((size_t)CC * CC * 2);
  _Float16*  w_laq   = (_Float16*)carve((size_t)CC * CC * 2);
  _Float16*  w_lak   = (_Float16*)carve((size_t)CC * CC * 2);
  _Float16*  w_lav   = (_Float16*)carve((size_t)CC * CC * 2);
  _Float16*  w_lao   = (_Float16*)carve((size_t)CC * CC * 2);
  _Float16*  xq_h    = (_Float16*)carve((size_t)INST * HWSZ * CC * 2);
  float*     q_f32   = (float*)carve((size_t)INST * HWSZ * CC * 4);
  float*     shading = (float*)carve((size_t)7 * HWSZ * CC * 4);
  _Float16*  ori_h   = (_Float16*)carve((size_t)HWSZ * CC * 2);
  float*     scale8  = (float*)carve((size_t)8 * HWSZ * 4);
  float*     scale30 = (float*)carve((size_t)NUMPRO * HWSZ * 4);
  float*     x_ca    = (float*)carve((size_t)NUMPRO * HWSZ * 4);
  float*     chmean  = (float*)carve(NUMPRO * 4);
  float*     chmax   = (float*)carve(NUMPRO * 4);
  float*     casig   = (float*)carve(NUMPRO * 4);
  float*     compmax = (float*)carve((size_t)HWSZ * 4);
  float*     compmean= (float*)carve((size_t)HWSZ * 4);
  float*     sigy    = (float*)carve((size_t)HWSZ * 4);
  int*       invp    = (int*)carve((NUMPRO - 1) * 4);
  float*     wfin    = (float*)carve((size_t)7 * HWSZ * 4);
  // aliases into dead regions (stream-ordered reuse)
  _Float16*  attn_ea = xq_h;                                  // dead after q GEMM
  _Float16*  lq_h    = (_Float16*)q_f32;                      // q_f32 dead after EA attn
  _Float16*  lk_h    = lq_h + (size_t)HWSZ * CC;
  _Float16*  lv_h    = lk_h + (size_t)HWSZ * CC;
  _Float16*  laattn  = lv_h + (size_t)HWSZ * CC;

  const int TPB = 256;
  const size_t GEMM_LDS = 16384;
  auto blocks = [](size_t n, int t) { return (unsigned)((n + t - 1) / t); };

  // 1. resizes
  k_resize<<<blocks((size_t)INST * 4096, TPB), TPB, 0, stream>>>(guid, gm, INST);
  k_resize<<<blocks(4096, TPB), TPB, 0, stream>>>(supm, sup, 1);
  // 2. mig -> d_out
  k_mig<<<blocks((size_t)HWSZ * CC, TPB), TPB, 0, stream>>>(ca_x, gm, (float*)d_out);
  // 3. pos_net
  k_posnet<<<INST, TPB, 0, stream>>>(box, pn_w1, pn_b1, pn_w2, pn_b2, pn_w3, pn_b3, boxtok);
  // 4. EA k/v
  k_kv_ea<<<blocks((size_t)INST * 2 * CC, TPB), TPB, 0, stream>>>(pooler, boxtok, ea_k_w, ea_v_w, kea, vea);
  // 5. LayerNorm of image_token[:,1:] -> xq f16
  k_ln_f16<<<INST * HWSZ, TPB, 0, stream>>>(itok + (size_t)HWSZ * CC, ln_w, ln_b, xq_h);
  // 6. weight converts
  size_t wn = (size_t)CC * CC;
  k_cvt_h<<<blocks(wn, TPB), TPB, 0, stream>>>(ea_q_w, w_eaq, wn);
  k_cvt_h<<<blocks(wn, TPB), TPB, 0, stream>>>(ea_o_w, w_eao, wn);
  k_cvt_h<<<blocks(wn, TPB), TPB, 0, stream>>>(la_q_w, w_laq, wn);
  k_cvt_h<<<blocks(wn, TPB), TPB, 0, stream>>>(la_k_w, w_lak, wn);
  k_cvt_h<<<blocks(wn, TPB), TPB, 0, stream>>>(la_v_w, w_lav, wn);
  k_cvt_h<<<blocks(wn, TPB), TPB, 0, stream>>>(la_o_w, w_lao, wn);
  // 7. q = xq @ ea_q_w^T  (f32 out)
  {
    dim3 g(CC / 64, (INST * HWSZ) / 64);
    k_gemm_wmma<<<g, TPB, GEMM_LDS, stream>>>(xq_h, w_eaq, nullptr, q_f32, INST * HWSZ, CC, CC, 0);
  }
  // 8. EA attention (2 keys) -> attn_ea f16 (aliases xq)
  k_ea_attn<<<blocks((size_t)INST * HWSZ * HEADS, TPB), TPB, 0, stream>>>(q_f32, kea, vea, attn_ea);
  // 9. ea = attn @ ea_out_w^T + b -> shading rows [0, 6*4096)
  {
    dim3 g(CC / 64, (INST * HWSZ) / 64);
    k_gemm_wmma<<<g, TPB, GEMM_LDS, stream>>>(attn_ea, w_eao, ea_o_b, shading, INST * HWSZ, CC, CC, 0);
  }
  // 10. ea *= gm
  k_scale_gm<<<blocks((size_t)INST * HWSZ * CC, TPB), TPB, 0, stream>>>(shading, gm);
  // 11. ori f16
  k_cvt_h<<<blocks((size_t)HWSZ * CC, TPB), TPB, 0, stream>>>(itok, ori_h, (size_t)HWSZ * CC);
  // 12. lq/lk/lv (f16 out, into dead q_f32 region)
  {
    dim3 g(CC / 64, HWSZ / 64);
    k_gemm_wmma<<<g, TPB, GEMM_LDS, stream>>>(ori_h, w_laq, nullptr, lq_h, HWSZ, CC, CC, 1);
    k_gemm_wmma<<<g, TPB, GEMM_LDS, stream>>>(ori_h, w_lak, nullptr, lk_h, HWSZ, CC, CC, 1);
    k_gemm_wmma<<<g, TPB, GEMM_LDS, stream>>>(ori_h, w_lav, nullptr, lv_h, HWSZ, CC, CC, 1);
  }
  // 13. fused flash local attention
  {
    dim3 g(HWSZ / 64, HEADS);
    k_flash_la<<<g, TPB, 0, stream>>>(lq_h, lk_h, lv_h, gm, sup, laattn);
  }
  // 14. fusion = laattn @ la_out_w^T + b -> shading row block 6
  {
    dim3 g(CC / 64, HWSZ / 64);
    k_gemm_wmma<<<g, TPB, GEMM_LDS, stream>>>(laattn, w_lao, la_o_b,
                                              shading + (size_t)INST * HWSZ * CC, HWSZ, CC, CC, 0);
  }
  // 15. SAC / CBAM chain
  k_scale8<<<(8 * HWSZ) / 8, TPB, 0, stream>>>(shading, sac_w, sac_b, scale8);
  k_build30<<<blocks((size_t)NUMPRO * HWSZ, TPB), TPB, 0, stream>>>(scale8, order, scale30);
  k_chanreduce<<<NUMPRO, TPB, 0, stream>>>(scale30, chmean, chmax);
  k_cbam_mlp<<<1, 32, 0, stream>>>(chmean, chmax, cb_w1, cb_b1, cb_w2, cb_b2, casig);
  k_xca_comp<<<blocks(HWSZ, TPB), TPB, 0, stream>>>(scale30, casig, x_ca, compmax, compmean);
  k_conv7<<<blocks(HWSZ, TPB), TPB, 0, stream>>>(compmax, compmean, cb_scw, cb_bnw, cb_bnb, sigy);
  k_inv<<<1, 32, 0, stream>>>(order, invp);
  k_wfin<<<blocks(HWSZ, TPB), TPB, 0, stream>>>(x_ca, sigy, invp, wfin);
  // 16. out = mig + tanh(gamma) * sum_p shading[p]*wfin[p]
  k_final<<<blocks((size_t)HWSZ * CC, TPB), TPB, 0, stream>>>(shading, wfin, gamma, (float*)d_out);

  (void)in_sizes; (void)n_in; (void)out_size; (void)ws_size;
}
